// QValueNet_79242146611608
// MI455X (gfx1250) — matrix operand order repaired and checked by reading, not compiled, not run on hardware
//
#include <hip/hip_runtime.h>

// ---------------- problem constants (from reference) ----------------
#define NN   100000   // nodes
#define NE   1600000  // edges
#define IN_D 128      // input dim
#define HID  64       // hidden dim
#define OUT_D 2

typedef __attribute__((ext_vector_type(16))) _Float16 v16h;
typedef __attribute__((ext_vector_type(8)))  float    v8f;

// ---------------- workspace layout (bytes, 256B aligned) ----------------
constexpr size_t OFF_DEG   = 0;                        // 100000 f32  (deg -> dinv in place)
constexpr size_t OFF_WPACK = 400384;                   // 2 * 8192 f16 (hi, lo)
constexpr size_t OFF_HP    = OFF_WPACK + 32768;        // 100000*64 f32  h' = (x@W)*dinv
constexpr size_t OFF_ACC   = OFF_HP + (size_t)NN*HID*4;// 100000*64 f32  scatter accumulator

__device__ __forceinline__ float fatomic_add(float* p, float v) {
#if defined(__HIP_DEVICE_COMPILE__)
  return unsafeAtomicAdd(p, v);   // hardware global_atomic_add_f32
#else
  return 0.f;
#endif
}

// ---------------- JAX Threefry-2x32, key = jax.random.key(42) => (0,42) ----
__device__ __forceinline__ unsigned tf_rotl(unsigned x, int r) {
  return (x << r) | (x >> (32 - r));
}
__device__ __forceinline__ unsigned threefry_bits(unsigned i) {
  // jax threefry_random_bits: counts = iota(6.4M), split in halves;
  // out[i] (i<half) = x0 of block (i, i+half); else x1 of block (i-half, i)
  const unsigned HALF = (unsigned)(NN * HID) / 2u;
  unsigned x0, x1; bool first;
  if (i < HALF) { x0 = i; x1 = i + HALF; first = true; }
  else          { x0 = i - HALF; x1 = i; first = false; }
  const unsigned ks0 = 0u, ks1 = 42u, ks2 = 0u ^ 42u ^ 0x1BD11BDAu;
  x0 += ks0; x1 += ks1;
#define TFR(r) { x0 += x1; x1 = tf_rotl(x1, r); x1 ^= x0; }
  TFR(13) TFR(15) TFR(26) TFR(6)   x0 += ks1; x1 += ks2 + 1u;
  TFR(17) TFR(29) TFR(16) TFR(24)  x0 += ks2; x1 += ks0 + 2u;
  TFR(13) TFR(15) TFR(26) TFR(6)   x0 += ks0; x1 += ks1 + 3u;
  TFR(17) TFR(29) TFR(16) TFR(24)  x0 += ks1; x1 += ks2 + 4u;
  TFR(13) TFR(15) TFR(26) TFR(6)   x0 += ks2; x1 += ks0 + 5u;
#undef TFR
  return first ? x0 : x1;
}

// ---------------- degree / dinv ----------------
__global__ void k_init_deg(float* __restrict__ deg) {
  int i = blockIdx.x * blockDim.x + threadIdx.x;
  if (i < NN) deg[i] = 1.0f;                 // self-loop
}
__global__ void k_count_deg(const int* __restrict__ col, float* __restrict__ deg) {
  int e = blockIdx.x * blockDim.x + threadIdx.x;
  if (e < NE) fatomic_add(&deg[col[e]], 1.0f);
}
__global__ void k_make_dinv(float* __restrict__ deg) {
  int i = blockIdx.x * blockDim.x + threadIdx.x;
  if (i < NN) deg[i] = rsqrtf(deg[i]);       // deg >= 1 always
}

// ---------------- pack W_conv (f32 [128x64]) into WMMA B layout, f16 hi/lo --
// B 32x16 (16-bit): lane n<16 -> col n, K j<8 -> K=j, j>=8 -> K=16+(j-8) ; lane n+16 -> base 8
__global__ void k_pack_w(const float* __restrict__ W, _Float16* __restrict__ wp) {
  int i = blockIdx.x * blockDim.x + threadIdx.x;    // 0..8191
  if (i >= 8192) return;
  int j    = i & 15;
  int lane = (i >> 4) & 31;
  int st   = i >> 9;                                // (s*4 + t), 0..15
  int s = st >> 2, t = st & 3;
  int K = 32 * s + 8 * (lane >> 4) + ((j < 8) ? j : (j + 8));
  int N = 16 * t + (lane & 15);
  float f = W[K * HID + N];
  _Float16 h = (_Float16)f;
  wp[i]        = h;                                 // hi part
  wp[8192 + i] = (_Float16)(f - (float)h);          // lo residual
}

// ---------------- WMMA GEMM: h' = (x @ W_conv) * dinv[row] ----------------
// one wave -> 16 rows x 64 cols; block 256 = 8 waves = 128 rows
// K-step loop kept as a hardware loop (unroll 1) to avoid VGPR spills.
__global__ void k_gemm(const float* __restrict__ x, const _Float16* __restrict__ wp,
                       const float* __restrict__ dinv, float* __restrict__ hp) {
  const int lane  = threadIdx.x & 31;
  const int wave  = threadIdx.x >> 5;
  const int rbase = (blockIdx.x * 8 + wave) * 16;
  if (rbase >= NN) return;                          // wave-uniform exit
  const int lmod = lane & 15, lhi = lane >> 4;
  const _Float16* wlo = wp + 8192;

  v8f acc[4] = {{}, {}, {}, {}};
  int arow = rbase + lmod; if (arow >= NN) arow = NN - 1;   // clamp for A loads
  const float* xrow = x + (size_t)arow * IN_D + lhi * 8;

  #pragma unroll 1
  for (int s = 0; s < 4; ++s) {
    // A tile 16x32 f16 (ISA layout: lanes 0-15 K base 0, lanes 16-31 K base 8;
    // j<8 -> K=base+j, j>=8 -> K=base+16+(j-8))
    const float* xr = xrow + s * 32;
    v16h ah, al;
    #pragma unroll
    for (int j = 0; j < 16; ++j) {
      float f = xr[(j < 8) ? j : (j + 8)];
      _Float16 h = (_Float16)f;
      ah[j] = h;
      al[j] = (_Float16)(f - (float)h);
    }
    #pragma unroll
    for (int t = 0; t < 4; ++t) {
      const int tile = ((s * 4 + t) * 32 + lane) * 16;
      v16h bh = *(const v16h*)(wp  + tile);
      v16h bl = *(const v16h*)(wlo + tile);
      acc[t] = __builtin_amdgcn_wmma_f32_16x16x32_f16(false, ah, false, bh, (short)0, acc[t], false, false);
      acc[t] = __builtin_amdgcn_wmma_f32_16x16x32_f16(false, al, false, bh, (short)0, acc[t], false, false);
      acc[t] = __builtin_amdgcn_wmma_f32_16x16x32_f16(false, ah, false, bl, (short)0, acc[t], false, false);
    }
  }
  // C/D layout: lane -> N = lmod, VGPR r -> M = r + 8*lhi
  #pragma unroll
  for (int r = 0; r < 8; ++r) {
    int row = rbase + r + 8 * lhi;
    if (row < NN) {
      float dv = dinv[row];
      hp[(size_t)row * HID +  0 + lmod] = acc[0][r] * dv;
      hp[(size_t)row * HID + 16 + lmod] = acc[1][r] * dv;
      hp[(size_t)row * HID + 32 + lmod] = acc[2][r] * dv;
      hp[(size_t)row * HID + 48 + lmod] = acc[3][r] * dv;
    }
  }
}

// ---------------- edge scatter: acc[dst] += h'[src]  (one wave per edge) ----
__global__ void k_scatter(const int* __restrict__ src, const int* __restrict__ dst,
                          const float* __restrict__ hp, float* __restrict__ acc) {
  long long tid = (long long)blockIdx.x * blockDim.x + threadIdx.x;
  int e = (int)(tid >> 5);
  int lane = (int)(tid & 31);
  if (e >= NE) return;
  const float* hs = hp  + (size_t)src[e] * HID;
  float*       ad = acc + (size_t)dst[e] * HID;
  fatomic_add(&ad[lane],      hs[lane]);
  fatomic_add(&ad[lane + 32], hs[lane + 32]);
}

// ---------------- finalize: out = dropout(relu(dinv*(acc+h') + b)) @ W_lin + b_lin
// one wave per node; lane handles dims (lane, lane+32); wave-reduce 2 outputs
__global__ void k_finalize(const float* __restrict__ hp, const float* __restrict__ acc,
                           const float* __restrict__ dinv, const float* __restrict__ bconv,
                           const float* __restrict__ wlin, const float* __restrict__ blin,
                           float* __restrict__ out) {
  int lane = threadIdx.x & 31;
  int n = (blockIdx.x * blockDim.x + threadIdx.x) >> 5;
  if (n >= NN) return;
  float dv = dinv[n];
  float y0 = 0.f, y1 = 0.f;
  #pragma unroll
  for (int h2 = 0; h2 < 2; ++h2) {
    int d = lane + 32 * h2;
    size_t idx = (size_t)n * HID + d;
    float hc = dv * (acc[idx] + hp[idx]) + bconv[d];   // self-loop term + bias
    hc = fmaxf(hc, 0.f);                               // relu
    unsigned bits = threefry_bits((unsigned)(n * HID + d));
    float u = __uint_as_float((bits >> 9) | 0x3F800000u) - 1.0f;
    hc = (u < 0.5f) ? hc * 2.0f : 0.f;                 // dropout p=0.5, /keep
    y0 += hc * wlin[d * 2 + 0];
    y1 += hc * wlin[d * 2 + 1];
  }
  #pragma unroll
  for (int off = 16; off > 0; off >>= 1) {
    y0 += __shfl_down(y0, off, 32);
    y1 += __shfl_down(y1, off, 32);
  }
  if (lane == 0) {
    out[(size_t)n * 2 + 0] = y0 + blin[0];
    out[(size_t)n * 2 + 1] = y1 + blin[1];
  }
}

// ---------------- launcher ----------------
extern "C" void kernel_launch(void* const* d_in, const int* in_sizes, int n_in,
                              void* d_out, int out_size, void* d_ws, size_t ws_size,
                              hipStream_t stream) {
  const float* x      = (const float*)d_in[0];
  const int*   eidx   = (const int*)  d_in[1];   // [2, NE]: src rows then dst rows
  const float* W_conv = (const float*)d_in[2];
  const float* b_conv = (const float*)d_in[3];
  const float* W_lin  = (const float*)d_in[4];
  const float* b_lin  = (const float*)d_in[5];
  float* out = (float*)d_out;

  char* ws = (char*)d_ws;
  float*    deg = (float*)   (ws + OFF_DEG);
  _Float16* wp  = (_Float16*)(ws + OFF_WPACK);
  float*    hp  = (float*)   (ws + OFF_HP);
  float*    acc = (float*)   (ws + OFF_ACC);

  const int* src = eidx;
  const int* dst = eidx + NE;

  // zero the scatter accumulator (graph-capture-legal)
  hipMemsetAsync(acc, 0, (size_t)NN * HID * sizeof(float), stream);

  k_init_deg <<<(NN + 255) / 256, 256, 0, stream>>>(deg);
  k_count_deg<<<(NE + 255) / 256, 256, 0, stream>>>(dst, deg);
  k_make_dinv<<<(NN + 255) / 256, 256, 0, stream>>>(deg);

  k_pack_w<<<32, 256, 0, stream>>>(W_conv, wp);

  k_gemm<<<(NN + 127) / 128, 256, 0, stream>>>(x, wp, deg, hp);

  k_scatter<<<(int)(((long long)NE * 32 + 255) / 256), 256, 0, stream>>>(src, dst, hp, acc);

  k_finalize<<<(NN * 32 + 255) / 256, 256, 0, stream>>>(hp, acc, deg, b_conv, W_lin, b_lin, out);
}
